// OneStepGenerator_30915174596776
// MI455X (gfx1250) — matrix operand order repaired and checked by reading, not compiled, not run on hardware
//
#include <hip/hip_runtime.h>
#include <math.h>

typedef __attribute__((ext_vector_type(16))) _Float16 v16h;
typedef __attribute__((ext_vector_type(8)))  float    v8f;
typedef __attribute__((ext_vector_type(4)))  float    f4v;

#define ENC_DIM 1024
#define HIDDEN  512
#define TOK_DIM 256
#define BATCH   32
#define CAND    32768
#define K2_NITER 4   // candidate tiles per wave in gather kernel

// ---------------------------------------------------------------------------
// Kernel 1: h = gelu(enc @ W1^T + b1), one 16x16 output tile per 1-wave block.
// A = enc tile (16xK f16 frags), B[k,n] = W1[n0+n, k].
// ---------------------------------------------------------------------------
__global__ __launch_bounds__(32)
void mlp1_kernel(const float* __restrict__ enc, const float* __restrict__ W1,
                 const float* __restrict__ b1, float* __restrict__ h)
{
    const int lane = threadIdx.x & 31;
    const int half = lane >> 4;
    const int l15  = lane & 15;
    const int n0 = blockIdx.x * 16;
    const int m0 = blockIdx.y * 16;

    const float* arow = enc + (m0 + l15) * ENC_DIM;
    const float* brow = W1  + (n0 + l15) * ENC_DIM;
    const int akoff = half * 8;    // A: lanes>=16 carry K+8 within each 16-chunk
    const int bkoff = half * 8;    // B: lanes>=16 carry K+8 within each 16-chunk

    v8f acc = {};
    for (int k0 = 0; k0 < ENC_DIM; k0 += 32) {
        v16h a, b;
#pragma unroll
        for (int j = 0; j < 8; ++j) {
            a[j]     = (_Float16)arow[k0 + akoff + j];
            a[j + 8] = (_Float16)arow[k0 + 16 + akoff + j];
        }
#pragma unroll
        for (int j = 0; j < 8; ++j) {
            b[j]     = (_Float16)brow[k0 + bkoff + j];
            b[j + 8] = (_Float16)brow[k0 + 16 + bkoff + j];
        }
        acc = __builtin_amdgcn_wmma_f32_16x16x32_f16(false, a, false, b,
                                                     (short)0, acc, false, false);
    }
    const float bias = b1[n0 + l15];
#pragma unroll
    for (int v = 0; v < 8; ++v) {
        float x = acc[v] + bias;
        float g = 0.5f * x * (1.0f + erff(x * 0.70710678118654752f)); // exact GELU
        h[(m0 + half * 8 + v) * HIDDEN + (n0 + l15)] = g;
    }
}

// ---------------------------------------------------------------------------
// Kernel 2: q = h @ W2^T (no bias), one 16x16 tile per 1-wave block.
// ---------------------------------------------------------------------------
__global__ __launch_bounds__(32)
void mlp2_kernel(const float* __restrict__ h, const float* __restrict__ W2,
                 float* __restrict__ q)
{
    const int lane = threadIdx.x & 31;
    const int half = lane >> 4;
    const int l15  = lane & 15;
    const int n0 = blockIdx.x * 16;
    const int m0 = blockIdx.y * 16;

    const float* arow = h  + (m0 + l15) * HIDDEN;
    const float* brow = W2 + (n0 + l15) * HIDDEN;
    const int akoff = half * 8;
    const int bkoff = half * 8;

    v8f acc = {};
    for (int k0 = 0; k0 < HIDDEN; k0 += 32) {
        v16h a, b;
#pragma unroll
        for (int j = 0; j < 8; ++j) {
            a[j]     = (_Float16)arow[k0 + akoff + j];
            a[j + 8] = (_Float16)arow[k0 + 16 + akoff + j];
        }
#pragma unroll
        for (int j = 0; j < 8; ++j) {
            b[j]     = (_Float16)brow[k0 + bkoff + j];
            b[j + 8] = (_Float16)brow[k0 + 16 + bkoff + j];
        }
        acc = __builtin_amdgcn_wmma_f32_16x16x32_f16(false, a, false, b,
                                                     (short)0, acc, false, false);
    }
#pragma unroll
    for (int v = 0; v < 8; ++v)
        q[(m0 + half * 8 + v) * TOK_DIM + (n0 + l15)] = acc[v];
}

// ---------------------------------------------------------------------------
// Kernel 3: logits[b,c] = q[b] . tok_emb[cand[b,c]]  (memory-bound gather-dot)
// A-matrix = 16 gathered embedding rows (M) x 32 dims (K), f32->f16.
// B-matrix = q[b] replicated across all 16 N columns (compute redundant,
// memory optimal). 8 K-steps of v_wmma_f32_16x16x32_f16 per 16 candidates.
//
// L2 residency policy (MI455X: tok_emb 102MB fits 192MB L2, ~10.5x reuse):
//   - embedding loads: default RT (keep resident in L2)
//   - cand_tok loads:  NT (streamed once, don't evict embeddings)
//   - logits stores:   NT (streamed once, don't evict embeddings)
// ---------------------------------------------------------------------------
__global__ __launch_bounds__(256)
void gather_dot_kernel(const float* __restrict__ q,    // (B, TOK_DIM) from ws
                       const int*   __restrict__ cand, // (B, C)
                       const float* __restrict__ emb,  // (VOCAB, TOK_DIM)
                       float* __restrict__ out)        // (B, C)
{
    __shared__ float qs[TOK_DIM];
    const int b   = blockIdx.y;
    const int tid = threadIdx.x;
    qs[tid] = q[b * TOK_DIM + tid];   // blockDim.x == TOK_DIM == 256
    __syncthreads();

    const int lane = tid & 31;
    const int wave = tid >> 5;
    const int half = lane >> 4;
    const int l15  = lane & 15;
    const int akoff = half * 8;

    // Pre-build B fragments for all 8 K-steps (value depends only on K).
    v16h bq[8];
#pragma unroll
    for (int s = 0; s < 8; ++s) {
        const int kbase = s * 32 + half * 8;
#pragma unroll
        for (int j = 0; j < 8; ++j) {
            bq[s][j]     = (_Float16)qs[kbase + j];
            bq[s][j + 8] = (_Float16)qs[kbase + 16 + j];
        }
    }

    const long brow  = (long)b * CAND;
    const int  cbase = blockIdx.x * (8 * 16 * K2_NITER) + wave * (16 * K2_NITER);

    for (int it = 0; it < K2_NITER; ++it) {
        const int c0  = cbase + it * 16;
        const int row = __builtin_nontemporal_load(&cand[brow + c0 + l15]);
        const float* erow = emb + (long)row * TOK_DIM;

        if (it + 1 < K2_NITER) {   // prefetch next tile's rows (two 512B-apart
                                   // segments per row via the two half-waves)
            const int nrow = __builtin_nontemporal_load(&cand[brow + c0 + 16 + l15]);
            __builtin_prefetch(emb + (long)nrow * TOK_DIM + half * 128, 0, 1);
        }

        v8f acc = {};
#pragma unroll
        for (int s = 0; s < 8; ++s) {
            const int k0 = s * 32;
            const float4 x0 = ((const float4*)(erow + k0 + akoff))[0];
            const float4 x1 = ((const float4*)(erow + k0 + akoff))[1];
            const float4 y0 = ((const float4*)(erow + k0 + 16 + akoff))[0];
            const float4 y1 = ((const float4*)(erow + k0 + 16 + akoff))[1];
            v16h a;
            a[0]  = (_Float16)x0.x; a[1]  = (_Float16)x0.y;
            a[2]  = (_Float16)x0.z; a[3]  = (_Float16)x0.w;
            a[4]  = (_Float16)x1.x; a[5]  = (_Float16)x1.y;
            a[6]  = (_Float16)x1.z; a[7]  = (_Float16)x1.w;
            a[8]  = (_Float16)y0.x; a[9]  = (_Float16)y0.y;
            a[10] = (_Float16)y0.z; a[11] = (_Float16)y0.w;
            a[12] = (_Float16)y1.x; a[13] = (_Float16)y1.y;
            a[14] = (_Float16)y1.z; a[15] = (_Float16)y1.w;
            acc = __builtin_amdgcn_wmma_f32_16x16x32_f16(false, a, false, bq[s],
                                                         (short)0, acc, false, false);
        }

        // All 16 N columns hold identical dot products; lane 0 has M=0..7,
        // lane 16 has M=8..15. Two 16B non-temporal stores per producing lane.
        if (l15 == 0) {
            f4v lo = {acc[0], acc[1], acc[2], acc[3]};
            f4v hi = {acc[4], acc[5], acc[6], acc[7]};
            f4v* o = (f4v*)(out + brow + c0 + half * 8);
            __builtin_nontemporal_store(lo, o);
            __builtin_nontemporal_store(hi, o + 1);
        }
    }
}

// ---------------------------------------------------------------------------
extern "C" void kernel_launch(void* const* d_in, const int* in_sizes, int n_in,
                              void* d_out, int out_size, void* d_ws, size_t ws_size,
                              hipStream_t stream)
{
    (void)in_sizes; (void)n_in; (void)out_size; (void)ws_size;
    const float* enc  = (const float*)d_in[0];
    const int*   cand = (const int*)  d_in[1];
    const float* emb  = (const float*)d_in[2];
    const float* W1   = (const float*)d_in[3];
    const float* b1   = (const float*)d_in[4];
    const float* W2   = (const float*)d_in[5];
    float* out = (float*)d_out;

    float* h = (float*)d_ws;            // 32*512 f32
    float* q = h + BATCH * HIDDEN;      // 32*256 f32

    mlp1_kernel<<<dim3(HIDDEN / 16, BATCH / 16), 32, 0, stream>>>(enc, W1, b1, h);
    mlp2_kernel<<<dim3(TOK_DIM / 16, BATCH / 16), 32, 0, stream>>>(h, W2, q);
    gather_dot_kernel<<<dim3(CAND / (8 * 16 * K2_NITER), BATCH), 256, 0, stream>>>(
        q, cand, emb, out);
}
